// RefineAttenpooling_33354716020877
// MI455X (gfx1250) — matrix operand order repaired and checked by reading, not compiled, not run on hardware
//
#include <hip/hip_runtime.h>
#include <hip/hip_bf16.h>
#include <math.h>

// Problem constants (from reference): x [16, 512, 48, 48] fp32
#define B  16
#define C  512
#define N  2304          // 48*48
#define NT 144           // N / 16 column tiles
#define KB 16            // C / 32 k-blocks for 16x16x32 WMMA
#define TILE_BYTES 16384 // one jt tile of B operand: KB * 1KB

typedef __attribute__((ext_vector_type(16))) __bf16 v16bf;
typedef __attribute__((ext_vector_type(8)))  float  v8f;

// ---------------------------------------------------------------------------
// Kernel 1: per (b, n): att = softplus(sum_c x*wq), inv = 1/max(||x||_2, eps)
// Threads along n -> loads over c are fully coalesced (stride-1 in n).
// ---------------------------------------------------------------------------
__global__ __launch_bounds__(256) void k_stats(const float* __restrict__ x,
                                               const float* __restrict__ wq,
                                               float* __restrict__ att,
                                               float* __restrict__ inv) {
  int tid = blockIdx.x * 256 + threadIdx.x;   // = b*N + n
  int b = tid / N;
  int n = tid - b * N;
  const float* px = x + (size_t)b * C * N + n;
  float q = 0.f, ss = 0.f;
#pragma unroll 8
  for (int c = 0; c < C; ++c) {
    float v = px[(size_t)c * N];
    q  += v * wq[c];       // wq[c] is wave-uniform -> scalar load
    ss += v * v;
  }
  float a = (q > 20.f) ? q : log1pf(expf(q));   // softplus
  att[tid] = a;
  inv[tid] = 1.f / fmaxf(sqrtf(ss), 1e-12f);
}

// ---------------------------------------------------------------------------
// Kernel 2: write normalized x (bf16) pre-swizzled into WMMA operand layouts.
//   FA (A-operand, 16x32, lane=M%16): e<8 -> K=e, else K=e+8; +8 if lane>=16
//       index: ((b*KB+kb)*NT + it)*512 + lane*16 + e      (per-(it) blocks)
//   FB (B-operand, 32x16, lane=N%16): e<8 -> K=e, else K=e+8; +8 if lane>=16
//       index: ((b*NT+jt)*KB + kb)*512 + lane*16 + e      (TILE-CONTIGUOUS:
//       one jt tile = 16KB flat block -> trivial async LDS staging)
// ---------------------------------------------------------------------------
__global__ __launch_bounds__(256) void k_pack(const float* __restrict__ x,
                                              const float* __restrict__ inv,
                                              __bf16* __restrict__ FA,
                                              __bf16* __restrict__ FB) {
  size_t d = (size_t)blockIdx.x * 256 + threadIdx.x;
  int e  = (int)(d & 15);
  int L  = (int)((d >> 4) & 31);
  size_t t = d >> 9;                 // (b*KB + kb)*NT + tile
  int it = (int)(t % NT);
  size_t bk = t / NT;                // b*KB + kb
  int kb = (int)(bk % KB);
  int b  = (int)(bk / KB);

  int col = it * 16 + (L & 15);
  float iv = inv[(size_t)b * N + col];
  const float* px = x + (size_t)b * C * N;

  int kA = kb * 32 + ((e < 8) ? e : e + 8) + ((L >= 16) ? 8 : 0);
  int kB = kb * 32 + ((e < 8) ? e : e + 8) + ((L >= 16) ? 8 : 0);

  FA[d] = (__bf16)(px[(size_t)kA * N + col] * iv);
  size_t fb = (((size_t)b * NT + it) * KB + kb) * 512 + (size_t)L * 16 + e;
  FB[fb] = (__bf16)(px[(size_t)kB * N + col] * iv);
}

// ---------------------------------------------------------------------------
// Async global->LDS staging of one 16KB B tile (1024 x 16B chunks, 256 thr).
// CDNA5 GLOBAL_LOAD_ASYNC_TO_LDS_B128: VDST = LDS byte offset, VADDR = 64-bit
// global address; tracked by ASYNCcnt (s_wait_asynccnt).
// ---------------------------------------------------------------------------
__device__ __forceinline__ void stage_tile_async(char* lds_dst, const char* gsrc,
                                                 int tflat) {
#pragma unroll
  for (int i = 0; i < 4; ++i) {
    int off = tflat * 16 + i * 4096;
    unsigned ldsoff = (unsigned)(uintptr_t)(lds_dst + off);
    const char* g = gsrc + off;
    asm volatile("global_load_async_to_lds_b128 %0, %1, off"
                 :: "v"(ldsoff), "v"(g) : "memory");
  }
}

__device__ __forceinline__ void wait_async_all() {
  asm volatile("s_wait_asynccnt 0" ::: "memory");
}

// ---------------------------------------------------------------------------
// Kernel 3: fused  refine[n] = sum_m relu(F^T F)[n,m]^2 * att[m]
// WG = 8 waves, each owning one of 8 consecutive n-tiles of one batch; all
// share the jt-tile B panel staged in LDS (cuts L2 B-traffic 8x: 5.3GB->660MB).
// Double-buffered async prefetch of tile jt+1 overlaps the 16-WMMA chain.
// A (full K=512) stays resident in 128 VGPRs per wave. TWO independent D
// accumulators (even/odd kb) break the WMMA->WMMA RAW chain on the C operand
// so the XDL pipe can take back-to-back issues; partials are summed before
// the relu (which must see the total similarity). D layout: vgpr r ->
// M=r (lanes 0-15) / M=r+8 (lanes 16-31); fuse relu^2 * att into per-lane
// accumulators, one cross-lane reduction at the end. No atomics.
// ---------------------------------------------------------------------------
__global__ __launch_bounds__(256) void k_gemm(const __bf16* __restrict__ FA,
                                              const __bf16* __restrict__ FB,
                                              const float* __restrict__ att,
                                              float* __restrict__ refine) {
  __shared__ __align__(16) char bB[2][TILE_BYTES];

  int widx = threadIdx.x >> 5;
  int lane = threadIdx.x & 31;
  int b   = blockIdx.x / (NT / 8);
  int it  = (blockIdx.x % (NT / 8)) * 8 + widx;

  // resident A operand: 16 k-blocks x 16 bf16/lane = 128 VGPRs
  v16bf a[KB];
#pragma unroll
  for (int kb = 0; kb < KB; ++kb) {
    a[kb] = *(const v16bf*)(FA + (((size_t)b * KB + kb) * NT + it) * 512 + lane * 16);
  }

  float acc[8];
#pragma unroll
  for (int r = 0; r < 8; ++r) acc[r] = 0.f;

  const float* pat = att + (size_t)b * N + (lane & 15);
  const char*  fbb = (const char*)(FB + (size_t)b * NT * KB * 512);

  // prologue: stage tile 0 into buffer 0
  stage_tile_async(&bB[0][0], fbb, threadIdx.x);
  wait_async_all();
  __syncthreads();

  for (int jt = 0; jt < NT; ++jt) {
    int cur = jt & 1;
    if (jt + 1 < NT)   // prefetch next tile into the other buffer (async)
      stage_tile_async(&bB[cur ^ 1][0], fbb + (size_t)(jt + 1) * TILE_BYTES,
                       threadIdx.x);

    v8f d0 = {};       // even kb partial
    v8f d1 = {};       // odd  kb partial  (independent WMMA chains)
#pragma unroll
    for (int kb = 0; kb < KB; kb += 2) {
      v16bf bv0 = *(const v16bf*)(&bB[cur][(kb + 0) * 1024 + lane * 32]);
      v16bf bv1 = *(const v16bf*)(&bB[cur][(kb + 1) * 1024 + lane * 32]);
      d0 = __builtin_amdgcn_wmma_f32_16x16x32_bf16(
          false, a[kb + 0], false, bv0, (short)0, d0, false, false);
      d1 = __builtin_amdgcn_wmma_f32_16x16x32_bf16(
          false, a[kb + 1], false, bv1, (short)0, d1, false, false);
    }
    float am = pat[jt * 16];       // att[m0 + (lane%16)]
#pragma unroll
    for (int r = 0; r < 8; ++r) {
      float s = fmaxf(d0[r] + d1[r], 0.f);   // relu of TOTAL similarity
      acc[r] += s * s * am;                  // relu(sim)^2 * att[m]
    }

    wait_async_all();              // prefetch landed
    __syncthreads();               // everyone done reading bB[cur]
  }

  // reduce over m within each 16-lane half; lanes 0/16 hold M=r / M=r+8 sums
#pragma unroll
  for (int r = 0; r < 8; ++r) {
    float s = acc[r];
    s += __shfl_xor(s, 1, 32);
    s += __shfl_xor(s, 2, 32);
    s += __shfl_xor(s, 4, 32);
    s += __shfl_xor(s, 8, 32);
    if (lane == 0)  refine[(size_t)b * N + it * 16 + r]     = s;
    if (lane == 16) refine[(size_t)b * N + it * 16 + r + 8] = s;
  }
}

// ---------------------------------------------------------------------------
// Kernel 4: out[b,c] = (1/N) * sum_n x[b,c,n] * refine[b,n]  (wave per (b,c))
// ---------------------------------------------------------------------------
__global__ __launch_bounds__(256) void k_final(const float* __restrict__ x,
                                               const float* __restrict__ refine,
                                               float* __restrict__ out) {
  int wave = blockIdx.x * 8 + (threadIdx.x >> 5);   // = b*C + c
  int lane = threadIdx.x & 31;
  int b = wave / C;
  const float* px = x + (size_t)wave * N;
  const float* pr = refine + (size_t)b * N;
  float s = 0.f;
  for (int n = lane; n < N; n += 32) s += px[n] * pr[n];
  s += __shfl_xor(s, 16, 32);
  s += __shfl_xor(s, 8, 32);
  s += __shfl_xor(s, 4, 32);
  s += __shfl_xor(s, 2, 32);
  s += __shfl_xor(s, 1, 32);
  if (lane == 0) out[wave] = s * (1.0f / N);
}

// ---------------------------------------------------------------------------
// Host launch. d_ws layout (bytes):
//   [0)        att    : B*N*4   = 147456
//   [147456)   inv    : B*N*4   = 147456
//   [294912)   refine : B*N*4   = 147456
//   [442368)   FA     : B*C*N*2 = 37748736   (A-layout bf16, it-blocked)
//   [+FA]      FB     : B*C*N*2 = 37748736   (B-layout bf16, jt-tile-contig)
//   total ~72.4 MB
// ---------------------------------------------------------------------------
extern "C" void kernel_launch(void* const* d_in, const int* in_sizes, int n_in,
                              void* d_out, int out_size, void* d_ws, size_t ws_size,
                              hipStream_t stream) {
  const float* x  = (const float*)d_in[0];
  const float* wq = (const float*)d_in[1];
  float* out = (float*)d_out;

  char* ws = (char*)d_ws;
  float*  att    = (float*)(ws);
  float*  inv    = (float*)(ws + 147456);
  float*  refine = (float*)(ws + 2 * 147456);
  __bf16* FA     = (__bf16*)(ws + 3 * 147456);
  __bf16* FB     = FA + (size_t)B * C * N;

  k_stats<<<(B * N) / 256, 256, 0, stream>>>(x, wq, att, inv);
  k_pack <<<(int)(((size_t)B * C * N) / 256), 256, 0, stream>>>(x, inv, FA, FB);
  k_gemm <<<B * (NT / 8), 256, 0, stream>>>(FA, FB, att, refine);
  k_final<<<(B * C) / 8, 256, 0, stream>>>(x, refine, out);
}
